// TransformerBlock_78606491452232
// MI455X (gfx1250) — matrix operand order repaired and checked by reading, not compiled, not run on hardware
//
#include <hip/hip_runtime.h>
#include <hip/hip_bf16.h>
#include <stdint.h>

// ---------------------------------------------------------------------------
// MI455X (gfx1250) transformer block, bf16 WMMA everywhere (16x16x32 bf16 has
// 8x the K-depth of the f32 WMMA shape; the block is compute-bound: ~344
// GFLOP vs ~150MB HBM traffic, intermediates L2-resident in 192MB L2).
// v5: GEMM A-band staging uses GLOBAL_LOAD_ASYNC_TO_LDS_B128 (ASYNCcnt,
// s_wait_asynccnt) — DMA directly into LDS, overlapping the WMMA burst.
// The builtin takes v4i pointers (per hipcc diagnostic), AS1 src / AS3 dst.
// ---------------------------------------------------------------------------

typedef __bf16 bf16_t;
typedef __attribute__((ext_vector_type(16))) __bf16 v16bf;
typedef __attribute__((ext_vector_type(8)))  __bf16 v8bf;
typedef __attribute__((ext_vector_type(8)))  float  v8f;
typedef __attribute__((ext_vector_type(4)))  int    v4i_t;

#define D_MODEL  1024
#define N_HEADS  16
#define HEAD_DIM 64
#define D_FF     4096
#define SEQ      2048
#define BATCH    4
#define NTOK     (BATCH * SEQ)      /* 8192 */
#define THETA    10000.0f
#define EPS      1e-5f

#define AROW 72                     /* 64 k + 8 pad (keeps 16B alignment) */

#if __has_builtin(__builtin_amdgcn_global_load_async_to_lds_b128) && \
    __has_builtin(__builtin_amdgcn_s_wait_asynccnt)
#define HAVE_ASYNC_LDS 1
#else
#define HAVE_ASYNC_LDS 0
#endif

typedef __attribute__((address_space(1))) v4i_t* g_v4i;
typedef __attribute__((address_space(3))) v4i_t* l_v4i;

// ----------------------------- helpers -------------------------------------

__device__ __forceinline__ bf16_t f2bf(float f) {
  union { float f; uint32_t u; } a; a.f = f;
  uint32_t r = a.u + 0x7FFFu + ((a.u >> 16) & 1u);   // round to nearest even
  union { uint16_t u; bf16_t b; } o; o.u = (uint16_t)(r >> 16);
  return o.b;
}

// A fragment (16x32 bf16 MxK), contiguous k = 16h + s layout, from arbitrary-stride mem.
__device__ __forceinline__ v16bf load_a_frag(const bf16_t* base, int stride) {
  int lane = threadIdx.x & 31;
  const bf16_t* p = base + (size_t)(lane & 15) * stride + ((lane >> 4) << 4);
  v8bf lo = *(const v8bf*)(p);
  v8bf hi = *(const v8bf*)(p + 8);
  v16bf r;
#pragma unroll
  for (int i = 0; i < 8; ++i) { r[i] = lo[i]; r[i + 8] = hi[i]; }
  return r;
}

// A fragment from an LDS-staged 32x64 tile (row stride AROW).
__device__ __forceinline__ v16bf lds_a_frag(const bf16_t* tile, int rowbase,
                                            int ks) {
  int lane = threadIdx.x & 31;
  const bf16_t* p =
      tile + (size_t)(rowbase + (lane & 15)) * AROW + ks + ((lane >> 4) << 4);
  v8bf lo = *(const v8bf*)(p);
  v8bf hi = *(const v8bf*)(p + 8);
  v16bf r;
#pragma unroll
  for (int i = 0; i < 8; ++i) { r[i] = lo[i]; r[i + 8] = hi[i]; }
  return r;
}

// B fragment (32x16 bf16 KxN), element (k,n) at base[n*stride + k].
__device__ __forceinline__ v16bf load_b_frag(const bf16_t* base, int stride) {
  int lane = threadIdx.x & 31;
  return *(const v16bf*)(base + (size_t)(lane & 15) * stride +
                         ((lane >> 4) << 4));
}

__device__ __forceinline__ v8f wmma_bf16(v16bf a, v16bf b, v8f c) {
  return __builtin_amdgcn_wmma_f32_16x16x32_bf16(false, a, false, b, (short)0,
                                                 c, false, false);
}

// Stage a 32x64 bf16 A chunk into LDS: 256 threads x 16B = 4KB.
// Async DMA (global -> LDS, ASYNCcnt) when available; else VGPR round-trip.
__device__ __forceinline__ void stage_a(bf16_t* dst, const bf16_t* A, int row0,
                                        int kk, int K) {
  int row = threadIdx.x >> 3;
  int seg = (threadIdx.x & 7) << 3;
  const bf16_t* src = A + (size_t)(row0 + row) * K + kk + seg;
  bf16_t* d = dst + (size_t)row * AROW + seg;
#if HAVE_ASYNC_LDS
  __builtin_amdgcn_global_load_async_to_lds_b128(
      (g_v4i)(v4i_t*)(void*)src, (l_v4i)(v4i_t*)(void*)d, 0, 0);
#else
  v8bf v = *(const v8bf*)src;
  *(v8bf*)d = v;
#endif
}

// Drain this wave's outstanding async LDS transfers (before the barrier).
__device__ __forceinline__ void stage_commit() {
#if HAVE_ASYNC_LDS
  __builtin_amdgcn_s_wait_asynccnt(0);
#endif
}

// --------------------------- small kernels ----------------------------------

__global__ void cvt_f32_to_bf16(const float* __restrict__ src,
                                bf16_t* __restrict__ dst, int n) {
  int i = blockIdx.x * blockDim.x + threadIdx.x;
  if (i < n) dst[i] = f2bf(src[i]);
}

__global__ void rmsnorm_bf16(const float* __restrict__ x,
                             const float* __restrict__ w,
                             bf16_t* __restrict__ out) {
  __shared__ float red[256];
  int row = blockIdx.x;
  int t   = threadIdx.x;
  const float4 v = ((const float4*)(x + (size_t)row * D_MODEL))[t];
  red[t] = v.x * v.x + v.y * v.y + v.z * v.z + v.w * v.w;
  __syncthreads();
#pragma unroll
  for (int s = 128; s > 0; s >>= 1) {
    if (t < s) red[t] += red[t + s];
    __syncthreads();
  }
  float scale = rsqrtf(red[0] * (1.0f / D_MODEL) + EPS);
  const float4 wv = ((const float4*)w)[t];
  bf16_t* o = out + (size_t)row * D_MODEL + t * 4;
  o[0] = f2bf(v.x * scale * wv.x);
  o[1] = f2bf(v.y * scale * wv.y);
  o[2] = f2bf(v.z * scale * wv.z);
  o[3] = f2bf(v.w * scale * wv.w);
}

__global__ void rope_qk(bf16_t* __restrict__ q, bf16_t* __restrict__ kk,
                        const int* __restrict__ tpos) {
  int idx = blockIdx.x * blockDim.x + threadIdx.x;   // NTOK*512 pairs
  int tok = idx >> 9;
  int rem = idx & 511;
  int h   = rem >> 5;
  int i   = rem & 31;
  int t   = tpos[tok & (SEQ - 1)];
  float inv = __powf(THETA, -(float)i * (1.0f / 32.0f));
  float c, s;
  __sincosf((float)t * inv, &s, &c);
  size_t base = (size_t)tok * D_MODEL + h * HEAD_DIM + 2 * i;
  {
    float x1 = (float)q[base], x2 = (float)q[base + 1];
    q[base]     = f2bf(x1 * c - x2 * s);
    q[base + 1] = f2bf(x1 * s + x2 * c);
  }
  {
    float x1 = (float)kk[base], x2 = (float)kk[base + 1];
    kk[base]     = f2bf(x1 * c - x2 * s);
    kk[base + 1] = f2bf(x1 * s + x2 * c);
  }
}

// ------------------------------ GEMM ----------------------------------------
// Block (256 thr = 8 waves) computes 32 rows x 512 cols. The 32-row A band is
// staged in LDS (double buffered, async DMA, one barrier per 64-deep K chunk)
// and shared by all 8 waves; each wave owns a 32x64 output tile.
// N,K are compile-time so all fragment addresses are base + immediate.
enum { MODE_BF16 = 0, MODE_RESID_F32 = 1, MODE_VT = 2 };

template <int MODE, int N, int K>
__launch_bounds__(256, 2)
__global__ void gemm_bf16(const bf16_t* __restrict__ A,
                          const bf16_t* __restrict__ W,
                          const float* __restrict__ resid,
                          void* __restrict__ out, int M) {
  __shared__ bf16_t As[2][32 * AROW];

  constexpr int nb_n = N >> 9;                 // N / 512
  int row0 = (blockIdx.x / nb_n) * 32;
  int col0 = (blockIdx.x % nb_n) * 512 + (threadIdx.x >> 5) * 64;
  int lane   = threadIdx.x & 31;
  int n_lane = lane & 15;
  int kb16   = (lane >> 4) << 4;

  // Single per-lane B base; the 4 fragment rows are constant offsets j*16*K.
  const bf16_t* bbase = W + (size_t)(col0 + n_lane) * K + kb16;

  v8f acc[2][4] = {};

  stage_a(As[0], A, row0, 0, K);
  stage_commit();
  __syncthreads();

  int buf = 0;
  for (int kk = 0; kk < K; kk += 64) {
    if (kk + 64 < K) stage_a(As[buf ^ 1], A, row0, kk + 64, K);
    __builtin_prefetch(bbase + kk + 128, 0, 1);   // warm next weight chunk
#pragma unroll
    for (int ks = 0; ks < 64; ks += 32) {
      const bf16_t* bp = bbase + kk + ks;
      v16bf b0 = *(const v16bf*)(bp);
      v16bf b1 = *(const v16bf*)(bp + 16 * K);
      v16bf b2 = *(const v16bf*)(bp + 32 * K);
      v16bf b3 = *(const v16bf*)(bp + 48 * K);
      v16bf a0 = lds_a_frag(As[buf], 0, ks);
      v16bf a1 = lds_a_frag(As[buf], 16, ks);
      acc[0][0] = wmma_bf16(a0, b0, acc[0][0]);
      acc[0][1] = wmma_bf16(a0, b1, acc[0][1]);
      acc[0][2] = wmma_bf16(a0, b2, acc[0][2]);
      acc[0][3] = wmma_bf16(a0, b3, acc[0][3]);
      acc[1][0] = wmma_bf16(a1, b0, acc[1][0]);
      acc[1][1] = wmma_bf16(a1, b1, acc[1][1]);
      acc[1][2] = wmma_bf16(a1, b2, acc[1][2]);
      acc[1][3] = wmma_bf16(a1, b3, acc[1][3]);
    }
    stage_commit();
    __syncthreads();
    buf ^= 1;
  }

  int mbase = (lane >> 4) << 3;
#pragma unroll
  for (int mt = 0; mt < 2; ++mt) {
#pragma unroll
    for (int j = 0; j < 4; ++j) {
#pragma unroll
      for (int r = 0; r < 8; ++r) {
        int row = row0 + mt * 16 + mbase + r;
        int col = col0 + j * 16 + n_lane;
        size_t idx = (size_t)row * N + col;
        if (MODE == MODE_BF16) {
          ((bf16_t*)out)[idx] = f2bf(acc[mt][j][r]);
        } else if (MODE == MODE_RESID_F32) {
          ((float*)out)[idx] = resid[idx] + acc[mt][j][r];
        } else {  // MODE_VT: vt[b][h][d][t]
          int b = row >> 11, t = row & (SEQ - 1);
          int h = col >> 6,  d = col & 63;
          ((bf16_t*)out)[((size_t)(b * N_HEADS + h) * HEAD_DIM + d) * SEQ + t] =
              f2bf(acc[mt][j][r]);
        }
      }
    }
  }
}

// Fused SwiGLU up-projection: act = silu(h2@w1^T) * (h2@w3^T). Same
// cooperative tiling; the two weight streams are processed back-to-back so
// only 4 B fragments are live at a time.
template <int N, int K>
__launch_bounds__(256, 2)
__global__ void gemm_swiglu(const bf16_t* __restrict__ A,
                            const bf16_t* __restrict__ W1,
                            const bf16_t* __restrict__ W3,
                            bf16_t* __restrict__ out, int M) {
  __shared__ bf16_t As[2][32 * AROW];

  constexpr int nb_n = N >> 9;
  int row0 = (blockIdx.x / nb_n) * 32;
  int col0 = (blockIdx.x % nb_n) * 512 + (threadIdx.x >> 5) * 64;
  int lane   = threadIdx.x & 31;
  int n_lane = lane & 15;
  int kb16   = (lane >> 4) << 4;

  size_t boff = (size_t)(col0 + n_lane) * K + kb16;
  const bf16_t* b1base = W1 + boff;
  const bf16_t* b3base = W3 + boff;

  v8f acc_a[2][4] = {};
  v8f acc_g[2][4] = {};

  stage_a(As[0], A, row0, 0, K);
  stage_commit();
  __syncthreads();

  int buf = 0;
  for (int kk = 0; kk < K; kk += 64) {
    if (kk + 64 < K) stage_a(As[buf ^ 1], A, row0, kk + 64, K);
    __builtin_prefetch(b1base + kk + 128, 0, 1);
    __builtin_prefetch(b3base + kk + 128, 0, 1);
#pragma unroll
    for (int ks = 0; ks < 64; ks += 32) {
      v16bf a0 = lds_a_frag(As[buf], 0, ks);
      v16bf a1 = lds_a_frag(As[buf], 16, ks);
      {
        const bf16_t* bp = b1base + kk + ks;
        v16bf b0 = *(const v16bf*)(bp);
        v16bf b1 = *(const v16bf*)(bp + 16 * K);
        v16bf b2 = *(const v16bf*)(bp + 32 * K);
        v16bf b3 = *(const v16bf*)(bp + 48 * K);
        acc_a[0][0] = wmma_bf16(a0, b0, acc_a[0][0]);
        acc_a[0][1] = wmma_bf16(a0, b1, acc_a[0][1]);
        acc_a[0][2] = wmma_bf16(a0, b2, acc_a[0][2]);
        acc_a[0][3] = wmma_bf16(a0, b3, acc_a[0][3]);
        acc_a[1][0] = wmma_bf16(a1, b0, acc_a[1][0]);
        acc_a[1][1] = wmma_bf16(a1, b1, acc_a[1][1]);
        acc_a[1][2] = wmma_bf16(a1, b2, acc_a[1][2]);
        acc_a[1][3] = wmma_bf16(a1, b3, acc_a[1][3]);
      }
      {
        const bf16_t* bp = b3base + kk + ks;
        v16bf b0 = *(const v16bf*)(bp);
        v16bf b1 = *(const v16bf*)(bp + 16 * K);
        v16bf b2 = *(const v16bf*)(bp + 32 * K);
        v16bf b3 = *(const v16bf*)(bp + 48 * K);
        acc_g[0][0] = wmma_bf16(a0, b0, acc_g[0][0]);
        acc_g[0][1] = wmma_bf16(a0, b1, acc_g[0][1]);
        acc_g[0][2] = wmma_bf16(a0, b2, acc_g[0][2]);
        acc_g[0][3] = wmma_bf16(a0, b3, acc_g[0][3]);
        acc_g[1][0] = wmma_bf16(a1, b0, acc_g[1][0]);
        acc_g[1][1] = wmma_bf16(a1, b1, acc_g[1][1]);
        acc_g[1][2] = wmma_bf16(a1, b2, acc_g[1][2]);
        acc_g[1][3] = wmma_bf16(a1, b3, acc_g[1][3]);
      }
    }
    stage_commit();
    __syncthreads();
    buf ^= 1;
  }

  int mbase = (lane >> 4) << 3;
#pragma unroll
  for (int mt = 0; mt < 2; ++mt) {
#pragma unroll
    for (int j = 0; j < 4; ++j) {
#pragma unroll
      for (int r = 0; r < 8; ++r) {
        int row = row0 + mt * 16 + mbase + r;
        int col = col0 + j * 16 + n_lane;
        float av = acc_a[mt][j][r];
        float gv = acc_g[mt][j][r];
        float silu = av * (1.0f / (1.0f + __expf(-av)));
        out[(size_t)row * N + col] = f2bf(silu * gv);
      }
    }
  }
}

// --------------------------- flash attention --------------------------------
__launch_bounds__(256, 2)
__global__ void flash_attn(const bf16_t* __restrict__ q,
                           const bf16_t* __restrict__ k,
                           const bf16_t* __restrict__ vt,
                           bf16_t* __restrict__ ctx) {
  __shared__ bf16_t pbuf[8][16 * 32];
  int warp = threadIdx.x >> 5;
  int lane = threadIdx.x & 31;
  int wid  = blockIdx.x * 8 + warp;
  int qt = wid & 127;
  int bh = wid >> 7;
  int b = bh >> 4, h = bh & 15;

  const bf16_t* qbase =
      q + (size_t)(b * SEQ + qt * 16) * D_MODEL + h * HEAD_DIM;
  v16bf aQ0 = load_a_frag(qbase, D_MODEL);
  v16bf aQ1 = load_a_frag(qbase + 32, D_MODEL);

  float m_i[8], l_i[8];
  v8f o[4] = {};
#pragma unroll
  for (int r = 0; r < 8; ++r) { m_i[r] = -3.0e38f; l_i[r] = 0.0f; }

  int n_lane = lane & 15;
  int mbase  = (lane >> 4) << 3;
  bf16_t* pw = pbuf[warp];
  int nkb = ((qt * 16 + 15) >> 5) + 1;

  for (int kb = 0; kb < nkb; ++kb) {
    int k0 = kb * 32;
    const bf16_t* kbase =
        k + (size_t)(b * SEQ + k0) * D_MODEL + h * HEAD_DIM;
    __builtin_prefetch(kbase + (size_t)32 * D_MODEL, 0, 1);
    v16bf kb00 = load_b_frag(kbase, D_MODEL);
    v16bf kb01 = load_b_frag(kbase + 32, D_MODEL);
    v16bf kb10 = load_b_frag(kbase + (size_t)16 * D_MODEL, D_MODEL);
    v16bf kb11 = load_b_frag(kbase + (size_t)16 * D_MODEL + 32, D_MODEL);
    v8f s0 = {}, s1 = {};
    s0 = wmma_bf16(aQ0, kb00, s0);
    s1 = wmma_bf16(aQ0, kb10, s1);
    s0 = wmma_bf16(aQ1, kb01, s0);
    s1 = wmma_bf16(aQ1, kb11, s1);

    float mnew[8], psum[8];
#pragma unroll
    for (int r = 0; r < 8; ++r) {
      int row = qt * 16 + mbase + r;
      float v0 = s0[r] * 0.125f;
      float v1 = s1[r] * 0.125f;
      if (k0 + n_lane > row)      v0 = -3.0e38f;
      if (k0 + 16 + n_lane > row) v1 = -3.0e38f;
      float mx = fmaxf(v0, v1);
#pragma unroll
      for (int off = 1; off < 16; off <<= 1)
        mx = fmaxf(mx, __shfl_xor(mx, off, 32));
      mnew[r] = fmaxf(m_i[r], mx);
      float p0 = __expf(v0 - mnew[r]);
      float p1 = __expf(v1 - mnew[r]);
      s0[r] = p0; s1[r] = p1;
      float sum = p0 + p1;
#pragma unroll
      for (int off = 1; off < 16; off <<= 1)
        sum += __shfl_xor(sum, off, 32);
      psum[r] = sum;
    }
#pragma unroll
    for (int r = 0; r < 8; ++r) {
      float sc = __expf(m_i[r] - mnew[r]);
      l_i[r] = l_i[r] * sc + psum[r];
      m_i[r] = mnew[r];
      o[0][r] *= sc; o[1][r] *= sc; o[2][r] *= sc; o[3][r] *= sc;
    }

#pragma unroll
    for (int r = 0; r < 8; ++r) {
      pw[(mbase + r) * 32 + n_lane]      = f2bf(s0[r]);
      pw[(mbase + r) * 32 + 16 + n_lane] = f2bf(s1[r]);
    }
    asm volatile("s_wait_dscnt 0" ::: "memory");
    v16bf aP = load_a_frag(pw, 32);

    const bf16_t* vb = vt + (size_t)bh * HEAD_DIM * SEQ + k0;
    v16bf vb0 = load_b_frag(vb, SEQ);
    v16bf vb1 = load_b_frag(vb + (size_t)16 * SEQ, SEQ);
    v16bf vb2 = load_b_frag(vb + (size_t)32 * SEQ, SEQ);
    v16bf vb3 = load_b_frag(vb + (size_t)48 * SEQ, SEQ);
    o[0] = wmma_bf16(aP, vb0, o[0]);
    o[1] = wmma_bf16(aP, vb1, o[1]);
    o[2] = wmma_bf16(aP, vb2, o[2]);
    o[3] = wmma_bf16(aP, vb3, o[3]);
  }

  bf16_t* cb = ctx + (size_t)(b * SEQ + qt * 16) * D_MODEL + h * HEAD_DIM;
#pragma unroll
  for (int j = 0; j < 4; ++j) {
#pragma unroll
    for (int r = 0; r < 8; ++r) {
      float inv = 1.0f / l_i[r];
      cb[(size_t)(mbase + r) * D_MODEL + j * 16 + n_lane] =
          f2bf(o[j][r] * inv);
    }
  }
}

// ------------------------------ launch --------------------------------------

extern "C" void kernel_launch(void* const* d_in, const int* in_sizes, int n_in,
                              void* d_out, int out_size, void* d_ws,
                              size_t ws_size, hipStream_t stream) {
  const float* x    = (const float*)d_in[0];
  const int*   tpos = (const int*)d_in[1];
  const float* ln1  = (const float*)d_in[2];
  const float* qw   = (const float*)d_in[3];
  const float* kw   = (const float*)d_in[4];
  const float* vw   = (const float*)d_in[5];
  const float* ow   = (const float*)d_in[6];
  const float* ln2  = (const float*)d_in[7];
  const float* w1   = (const float*)d_in[8];
  const float* w2   = (const float*)d_in[9];
  const float* w3   = (const float*)d_in[10];
  float* out = (float*)d_out;

  char* ws = (char*)d_ws;
  auto alloc = [&](size_t bytes) -> char* {
    char* p = ws;
    ws += (bytes + 255) & ~(size_t)255;
    return p;
  };

  const size_t DD = (size_t)D_MODEL * D_MODEL;
  const size_t DF = (size_t)D_FF * D_MODEL;
  const size_t TD = (size_t)NTOK * D_MODEL;
  const size_t TF = (size_t)NTOK * D_FF;

  bf16_t* qwb = (bf16_t*)alloc(DD * 2);
  bf16_t* kwb = (bf16_t*)alloc(DD * 2);
  bf16_t* vwb = (bf16_t*)alloc(DD * 2);
  bf16_t* owb = (bf16_t*)alloc(DD * 2);
  bf16_t* w1b = (bf16_t*)alloc(DF * 2);
  bf16_t* w3b = (bf16_t*)alloc(DF * 2);
  bf16_t* w2b = (bf16_t*)alloc(DF * 2);
  bf16_t* hbf = (bf16_t*)alloc(TD * 2);
  bf16_t* qbf = (bf16_t*)alloc(TD * 2);
  bf16_t* kbf = (bf16_t*)alloc(TD * 2);
  bf16_t* vtb = (bf16_t*)alloc(TD * 2);   // V transposed: [b][h][d][t]
  bf16_t* ctx = (bf16_t*)alloc(TD * 2);
  float*  x1  = (float*)alloc(TD * 4);
  bf16_t* h2b = (bf16_t*)alloc(TD * 2);
  bf16_t* act = (bf16_t*)alloc(TF * 2);

  cvt_f32_to_bf16<<<(int)(DD / 256), 256, 0, stream>>>(qw, qwb, (int)DD);
  cvt_f32_to_bf16<<<(int)(DD / 256), 256, 0, stream>>>(kw, kwb, (int)DD);
  cvt_f32_to_bf16<<<(int)(DD / 256), 256, 0, stream>>>(vw, vwb, (int)DD);
  cvt_f32_to_bf16<<<(int)(DD / 256), 256, 0, stream>>>(ow, owb, (int)DD);
  cvt_f32_to_bf16<<<(int)(DF / 256), 256, 0, stream>>>(w1, w1b, (int)DF);
  cvt_f32_to_bf16<<<(int)(DF / 256), 256, 0, stream>>>(w3, w3b, (int)DF);
  cvt_f32_to_bf16<<<(int)(DF / 256), 256, 0, stream>>>(w2, w2b, (int)DF);

  rmsnorm_bf16<<<NTOK, 256, 0, stream>>>(x, ln1, hbf);

  // Cooperative GEMM grid: (M/32) * (N/512) blocks of 256 threads.
  int blk_d = (NTOK / 32) * (D_MODEL / 512);        // 512
  gemm_bf16<MODE_BF16, D_MODEL, D_MODEL>
      <<<blk_d, 256, 0, stream>>>(hbf, qwb, nullptr, qbf, NTOK);
  gemm_bf16<MODE_BF16, D_MODEL, D_MODEL>
      <<<blk_d, 256, 0, stream>>>(hbf, kwb, nullptr, kbf, NTOK);
  gemm_bf16<MODE_VT, D_MODEL, D_MODEL>
      <<<blk_d, 256, 0, stream>>>(hbf, vwb, nullptr, vtb, NTOK);

  rope_qk<<<(NTOK * 512) / 256, 256, 0, stream>>>(qbf, kbf, tpos);

  flash_attn<<<(BATCH * N_HEADS * (SEQ / 16)) / 8, 256, 0, stream>>>(qbf, kbf,
                                                                     vtb, ctx);

  gemm_bf16<MODE_RESID_F32, D_MODEL, D_MODEL>
      <<<blk_d, 256, 0, stream>>>(ctx, owb, x, x1, NTOK);

  rmsnorm_bf16<<<NTOK, 256, 0, stream>>>(x1, ln2, h2b);

  int blk_ff = (NTOK / 32) * (D_FF / 512);          // 2048
  gemm_swiglu<D_FF, D_MODEL>
      <<<blk_ff, 256, 0, stream>>>(h2b, w1b, w3b, act, NTOK);

  gemm_bf16<MODE_RESID_F32, D_MODEL, D_FF>
      <<<blk_d, 256, 0, stream>>>(act, w2b, x1, out, NTOK);
}
